// Experts_41051297415846
// MI455X (gfx1250) — matrix operand order repaired and checked by reading, not compiled, not run on hardware
//
#include <hip/hip_runtime.h>

// Grouped per-expert SwiGLU FFN for MI455X (gfx1250, wave32, WMMA).
// E=8, T=256, D=2048, H=8192, fp32 in/out.
// Pass 1: mid[e,t,h] = silu(x@w1) * (x@w3)   (f16 intermediate in d_ws)
// Pass 2: out[e,t,d] = mid @ w2^T
//
// LDS tiles are FRAGMENT-PACKED: [strip][lane][16 halfs] -> every fragment
// load is 2x ds_load_b128, every staging write is 1x ds_store_b128.
// All fragments are preloaded before the WMMA chain so the 8 v_wmma issue
// back-to-back behind a single s_wait_dscnt.

typedef __attribute__((ext_vector_type(16))) _Float16 v16h;
typedef __attribute__((ext_vector_type(8)))  float    v8f;
typedef __attribute__((ext_vector_type(4)))  int      v4i;

#define NE 8
#define NT 256
#define ND 2048
#define NH 8192

#define BM 128      // token rows per workgroup (8 waves x 16-row strips)
#define BN 64       // output cols per workgroup (4 x 16-col strips)
#define BK 32       // K step == WMMA K for f16

// Packed tile sizes (halfs): strip -> 32 lanes x 16 halfs = 512
#define A_TILE_H (8 * 512)   // BM/16 strips
#define B_TILE_H (4 * 512)   // BN/16 strips

#if defined(__has_builtin)
#if __has_builtin(__builtin_amdgcn_global_load_async_to_lds_b128)
#define USE_ASYNC_LDS 1
#endif
#endif

// Builtin signature (probe-learned): (v4i AS1*, v4i AS3*, imm offset, cpol)
typedef __attribute__((address_space(1))) v4i gv4i_t;   // global b128
typedef __attribute__((address_space(3))) v4i lv4i_t;   // LDS b128

// ---- packed-layout mapping (CDNA5 ISA 7.12.2 wave32 layouts) ----
// A 16x32 f16: lane<16 holds row=lane, elems {K0..7, K16..23};
//              lane>=16 holds row=lane-16, elems {K8..15, K24..31}.
// Staging octet (row r, k in [c,c+8)), c in {0,8,16,24}:
//   lane = (r&15) + (((c>>3)&1)<<4),  j0 = (c & 16) >> 1
// B 32x16 f16: lane<16 holds col=lane, elems {K0..7, K16..23}; lane>=16 col=lane-16, {K8..15, K24..31}.
// Staging octet (col n, k in [k0,k0+8)), k0 in {0,8,16,24}:
//   lane = (n&15) + (((k0>>3)&1)<<4), j0 = (k0 & 16) >> 1

// ---- Pass 1: fused gate+up + SwiGLU -> f16 mid ----
// grid: (NH/BN, NT/BM, NE), block 256 (8 waves)
__global__ __launch_bounds__(256)
void swiglu_gate_up_kernel(const float* __restrict__ x,
                           const float* __restrict__ w1,
                           const float* __restrict__ w3,
                           _Float16* __restrict__ mid) {
    __shared__ __align__(32) _Float16 lXp [A_TILE_H];
    __shared__ __align__(32) _Float16 lW1p[B_TILE_H];
    __shared__ __align__(32) _Float16 lW3p[B_TILE_H];

    const int e    = blockIdx.z;
    const int nBlk = blockIdx.x * BN;   // over H
    const int mBlk = blockIdx.y * BM;   // over T
    const int tid  = threadIdx.x;
    const int lane = tid & 31;
    const int wave = tid >> 5;          // row strip 0..7

    const float* xe  = x  + (size_t)e * NT * ND + (size_t)mBlk * ND;
    const float* w1e = w1 + (size_t)e * ND * NH + nBlk;
    const float* w3e = w3 + (size_t)e * ND * NH + nBlk;

    // Staging coordinates (constant across K loop)
    const int br  = (tid * 2) >> 2;            // A row handled: tid/2
    const int bc0 = ((tid * 2) & 3) * 8;       // first octet (0 or 16)
    const int wn  = tid & 63;                  // B column
    const int wk0 = (tid >> 6) * 8;            // B k-octet
    const int wlane = (wn & 15) + (((wk0 >> 3) & 1) << 4);
    const int wj0   = (wk0 & 16) >> 1;
    _Float16* w1dst = &lW1p[(wn >> 4) * 512 + wlane * 16 + wj0];
    _Float16* w3dst = &lW3p[(wn >> 4) * 512 + wlane * 16 + wj0];

    v8f acc1[4] = {};
    v8f acc3[4] = {};

    for (int kb = 0; kb < ND; kb += BK) {
        // Stage X tile (BM x 32 f32 -> f16): 2 octets/thread, 1 b128 store each
#pragma unroll
        for (int uu = 0; uu < 2; ++uu) {
            const int c = bc0 + uu * 8;                      // 0,8,16,24
            const float* src = xe + (size_t)br * ND + kb + c;
            const int alane = (br & 15) + (((c >> 3) & 1) << 4);
            const int aj0   = (c & 16) >> 1;
            _Float16* dst = &lXp[(br >> 4) * 512 + alane * 16 + aj0];
#pragma unroll
            for (int j = 0; j < 8; ++j) dst[j] = (_Float16)src[j];
        }
        // Stage W1/W3 tiles: per-k reads (coalesced in n across lanes),
        // one contiguous b128 LDS store each
#pragma unroll
        for (int j = 0; j < 8; ++j)
            w1dst[j] = (_Float16)w1e[(size_t)(kb + wk0 + j) * NH + wn];
#pragma unroll
        for (int j = 0; j < 8; ++j)
            w3dst[j] = (_Float16)w3e[(size_t)(kb + wk0 + j) * NH + wn];
        __syncthreads();

        // Prefetch next weight K-tile (global_prefetch_b8)
        if (kb + BK < ND) {
            __builtin_prefetch(w1e + (size_t)(kb + BK + wk0) * NH + wn, 0, 1);
            __builtin_prefetch(w3e + (size_t)(kb + BK + wk0) * NH + wn, 0, 1);
        }

        // Preload ALL fragments (wide ds_load_b128s, one wait), then run the
        // 8-deep WMMA chain back-to-back on independent accumulators.
        const v16h af = *(const v16h*)&lXp[wave * 512 + lane * 16];
        v16h bf1[4], bf3[4];
#pragma unroll
        for (int s = 0; s < 4; ++s) {
            bf1[s] = *(const v16h*)&lW1p[s * 512 + lane * 16];
            bf3[s] = *(const v16h*)&lW3p[s * 512 + lane * 16];
        }
#pragma unroll
        for (int s = 0; s < 4; ++s) {
            acc1[s] = __builtin_amdgcn_wmma_f32_16x16x32_f16(
                false, af, false, bf1[s], (short)0, acc1[s], false, false);
            acc3[s] = __builtin_amdgcn_wmma_f32_16x16x32_f16(
                false, af, false, bf3[s], (short)0, acc3[s], false, false);
        }
        __syncthreads();
    }

    // Epilogue: silu(h1)*h3 -> f16 mid. C layout: VGPR i -> row i (+8 if lane>=16)
    _Float16* midp = mid + (size_t)e * NT * NH;
    const int rowB = mBlk + wave * 16 + ((lane < 16) ? 0 : 8);
    const int colB = nBlk + (lane & 15);
#pragma unroll
    for (int s = 0; s < 4; ++s) {
#pragma unroll
        for (int i = 0; i < 8; ++i) {
            const float g   = acc1[s][i];
            const float u   = acc3[s][i];
            const float sig = 1.0f / (1.0f + __expf(-g));
            midp[(size_t)(rowB + i) * NH + colB + 16 * s] = (_Float16)(g * sig * u);
        }
    }
}

// ---- Pass 2: down projection out = mid @ w2^T ----
// B[k=h, n=d] = w2[d*NH + h] (contiguous along k for fixed n)
// grid: (ND/BN, NT/BM, NE), block 256
__global__ __launch_bounds__(256)
void down_proj_kernel(const _Float16* __restrict__ mid,
                      const float* __restrict__ w2,
                      float* __restrict__ out) {
    __shared__ __align__(32) _Float16 lAp[A_TILE_H];
    __shared__ __align__(32) _Float16 lBp[B_TILE_H];

    const int e    = blockIdx.z;
    const int nBlk = blockIdx.x * BN;   // over D
    const int mBlk = blockIdx.y * BM;   // over T
    const int tid  = threadIdx.x;
    const int lane = tid & 31;
    const int wave = tid >> 5;

    const _Float16* ae  = mid + (size_t)e * NT * NH + (size_t)mBlk * NH;
    const float*    w2e = w2  + (size_t)e * ND * NH;

    const int br  = (tid * 2) >> 2;
    const int bc0 = ((tid * 2) & 3) * 8;
    const int wn  = tid & 63;
    const int wk0 = (tid >> 6) * 8;
    const int wlane = (wn & 15) + (((wk0 >> 3) & 1) << 4);
    const int wj0   = (wk0 & 16) >> 1;
    _Float16* bdst = &lBp[(wn >> 4) * 512 + wlane * 16 + wj0];

    v8f acc[4] = {};

    for (int kb = 0; kb < NH; kb += BK) {
        // Stage A tile (mid, already f16): pure 16B copies -> async DMA to LDS
#pragma unroll
        for (int uu = 0; uu < 2; ++uu) {
            const int c = bc0 + uu * 8;
            const _Float16* src = ae + (size_t)br * NH + kb + c;
            const int alane = (br & 15) + (((c >> 3) & 1) << 4);
            const int aj0   = (c & 16) >> 1;
            _Float16* dst = &lAp[(br >> 4) * 512 + alane * 16 + aj0];
#ifdef USE_ASYNC_LDS
            __builtin_amdgcn_global_load_async_to_lds_b128(
                (gv4i_t*)src, (lv4i_t*)dst, 0, 0);
#else
#pragma unroll
            for (int j = 0; j < 8; ++j) dst[j] = src[j];
#endif
        }
        // Stage B tile from w2: 8 contiguous f32 reads -> f16 -> one b128 store
        {
            const float* src = w2e + (size_t)(nBlk + wn) * NH + kb + wk0;
#pragma unroll
            for (int j = 0; j < 8; ++j) bdst[j] = (_Float16)src[j];
        }
#ifdef USE_ASYNC_LDS
#if __has_builtin(__builtin_amdgcn_s_wait_asynccnt)
        __builtin_amdgcn_s_wait_asynccnt(0);
#endif
#endif
        __syncthreads();

        if (kb + BK < NH) {
            __builtin_prefetch(w2e + (size_t)(nBlk + wn) * NH + kb + BK, 0, 1);
        }

        // Preload all fragments, then back-to-back WMMA chain
        const v16h af = *(const v16h*)&lAp[wave * 512 + lane * 16];
        v16h bf[4];
#pragma unroll
        for (int s = 0; s < 4; ++s)
            bf[s] = *(const v16h*)&lBp[s * 512 + lane * 16];
#pragma unroll
        for (int s = 0; s < 4; ++s) {
            acc[s] = __builtin_amdgcn_wmma_f32_16x16x32_f16(
                false, af, false, bf[s], (short)0, acc[s], false, false);
        }
        __syncthreads();
    }

    float* op      = out + (size_t)e * NT * ND;
    const int rowB = mBlk + wave * 16 + ((lane < 16) ? 0 : 8);
    const int colB = nBlk + (lane & 15);
#pragma unroll
    for (int s = 0; s < 4; ++s) {
#pragma unroll
        for (int i = 0; i < 8; ++i) {
            op[(size_t)(rowB + i) * ND + colB + 16 * s] = acc[s][i];
        }
    }
}

extern "C" void kernel_launch(void* const* d_in, const int* in_sizes, int n_in,
                              void* d_out, int out_size, void* d_ws, size_t ws_size,
                              hipStream_t stream) {
    const float* x  = (const float*)d_in[0];
    const float* w1 = (const float*)d_in[1];
    const float* w2 = (const float*)d_in[2];
    const float* w3 = (const float*)d_in[3];
    float* out      = (float*)d_out;
    _Float16* mid   = (_Float16*)d_ws;  // E*T*H f16 = 33.5 MB scratch

    dim3 g1(NH / BN, NT / BM, NE);
    swiglu_gate_up_kernel<<<g1, 256, 0, stream>>>(x, w1, w3, mid);

    dim3 g2(ND / BN, NT / BM, NE);
    down_proj_kernel<<<g2, 256, 0, stream>>>(mid, w2, out);
}
